// EinFFT_90263032693007
// MI455X (gfx1250) — matrix operand order repaired and checked by reading, not compiled, not run on hardware
//
#include <hip/hip_runtime.h>

// EinFFT forward for MI455X (gfx1250): token DFT (56x56 Cooley-Tukey) +
// block DFT-4 + complex block-diagonal MLP + inverse transforms.
// All dense matmuls run on v_wmma_f32_16x16x32_f16 (f16 operands, f32 acc).
// Intermediate buffers are row-padded (k up to 63) so epilogue stores are
// branch-free; twiddles and negated DFT matrices are staged in LDS.

typedef __attribute__((ext_vector_type(16))) _Float16 v16h;
typedef __attribute__((ext_vector_type(8)))  float    v8f;

#define PI_D 3.14159265358979323846

// padded plane: rows (k*56 + l) for k in [0,64), l in [0,56)  -> max 3583
#define PROWS 3584
#define TOKPAD 3584   // padded tokens per batch in bufB/bufC space

__device__ __forceinline__ v8f wmma_f16(v16h a, v16h b, v8f c) {
  return __builtin_amdgcn_wmma_f32_16x16x32_f16(false, a, false, b, (short)0, c,
                                                false, false);
}

__device__ __forceinline__ v8f zero8() {
  v8f z;
#pragma unroll
  for (int i = 0; i < 8; ++i) z[i] = 0.0f;
  return z;
}

// A fragment (16x32, f16): lane -> row = row0 + (lane&15);
// K pattern: k = k0 + 8*(lane>>4) + e + (e>=8 ? 8 : 0)
__device__ __forceinline__ v16h frag_a(const _Float16* S, int lda, int row0, int k0) {
  int lane = threadIdx.x & 31;
  const _Float16* row = S + (size_t)(row0 + (lane & 15)) * lda + k0 + ((lane >> 4) << 3);
  v16h f;
#pragma unroll
  for (int e = 0; e < 16; ++e) f[e] = row[e + ((e >= 8) ? 8 : 0)];
  return f;
}

// B fragment (32x16, f16): lane -> N = n0 + (lane&15);
// K pattern: k = k0 + 8*(lane>>4) + e + (e>=8 ? 8 : 0)
__device__ __forceinline__ v16h frag_b(const _Float16* S, int ldb, int k0, int n0) {
  int lane = threadIdx.x & 31;
  const _Float16* p = S + (size_t)(k0 + ((lane >> 4) << 3)) * ldb + n0 + (lane & 15);
  v16h f;
#pragma unroll
  for (int e = 0; e < 16; ++e) f[e] = p[(size_t)(e + ((e >= 8) ? 8 : 0)) * ldb];
  return f;
}

__device__ __forceinline__ float sshrink(float v) {
  return v > 0.01f ? v - 0.01f : (v < -0.01f ? v + 0.01f : 0.0f);
}

// ---------------------------------------------------------------------------
// K0: constants: padded 64x64 f16 DFT-56 matrices (fwd & inv), f32 twiddles,
// f16 copies of the MLP weights.
// ---------------------------------------------------------------------------
__global__ void k0_init(const float* __restrict__ w1, const float* __restrict__ w2,
                        _Float16* Wf_re, _Float16* Wf_im,
                        _Float16* Wi_re, _Float16* Wi_im,
                        float* TwF_re, float* TwF_im,
                        float* TwI_re, float* TwI_im,
                        _Float16* gW1h, _Float16* gW2h) {
  int i = blockIdx.x * 256 + threadIdx.x;
  if (i < 64 * 64) {
    int k = i >> 6, mm = i & 63;
    float re = 0.0f, imf = 0.0f, imi = 0.0f;
    if (k < 56 && mm < 56) {
      double th = 2.0 * PI_D * (double)(k * mm) / 56.0;
      re = (float)cos(th);
      imf = (float)(-sin(th));
      imi = (float)sin(th);
    }
    Wf_re[i] = (_Float16)re;  Wf_im[i] = (_Float16)imf;
    Wi_re[i] = (_Float16)re;  Wi_im[i] = (_Float16)imi;
  }
  if (i < 56 * 56) {
    int k = i / 56, l = i % 56;
    double th = 2.0 * PI_D * (double)(k * l) / 3136.0;
    float c = (float)cos(th), s = (float)sin(th);
    TwF_re[i] = c;  TwF_im[i] = -s;
    TwI_re[i] = c;  TwI_im[i] = s;
  }
  if (i < 2 * 4 * 128 * 128) {
    gW1h[i] = (_Float16)w1[i];
    gW2h[i] = (_Float16)w2[i];
  }
}

// ---------------------------------------------------------------------------
// K1: token DFT stage 1.  WG = (l, b).  out[k, r] = sum_m W56[k,m] * x[m,l,r]
// (real input -> 2 planes), twiddle epilogue.  bufA[(k*56+l)][b*512+c].
// ---------------------------------------------------------------------------
__global__ void k1_stage1(const float* __restrict__ x,
                          const _Float16* __restrict__ gWre, const _Float16* __restrict__ gWim,
                          const float* __restrict__ TwRe, const float* __restrict__ TwIm,
                          _Float16* __restrict__ bufA_re, _Float16* __restrict__ bufA_im) {
  extern __shared__ char smem[];
  _Float16* sB   = (_Float16*)smem;            // 64 x 512
  _Float16* sWre = sB + 64 * 512;              // 64 x 64
  _Float16* sWim = sWre + 64 * 64;
  float*    sTwR = (float*)(sWim + 64 * 64);   // 64
  float*    sTwI = sTwR + 64;
  int tid = threadIdx.x;
  int l = blockIdx.x, b = blockIdx.y;

  for (int idx = tid; idx < 64 * 512; idx += 256) {
    int m = idx >> 9, c = idx & 511;
    float v = 0.0f;
    if (m < 56) v = x[((size_t)b * 3136 + (size_t)m * 56 + l) * 512 + c];
    sB[idx] = (_Float16)v;
  }
  for (int idx = tid; idx < 64 * 64; idx += 256) {
    sWre[idx] = gWre[idx];
    sWim[idx] = gWim[idx];
  }
  if (tid < 64) {
    sTwR[tid] = (tid < 56) ? TwRe[tid * 56 + l] : 0.0f;
    sTwI[tid] = (tid < 56) ? TwIm[tid * 56 + l] : 0.0f;
  }
  __syncthreads();

  int lane = tid & 31, wave = tid >> 5;
  int colh = lane & 15, hi = lane >> 4;

  for (int kt = 0; kt < 4; ++kt) {
    v16h ar0 = frag_a(sWre, 64, kt * 16, 0);
    v16h ar1 = frag_a(sWre, 64, kt * 16, 32);
    v16h ai0 = frag_a(sWim, 64, kt * 16, 0);
    v16h ai1 = frag_a(sWim, 64, kt * 16, 32);
    for (int i = 0; i < 4; ++i) {
      int ct = wave * 4 + i;
      v16h b0 = frag_b(sB, 512, 0, ct * 16);
      v16h b1 = frag_b(sB, 512, 32, ct * 16);
      v8f cr = zero8(), ci = zero8();
      cr = wmma_f16(ar0, b0, cr);
      cr = wmma_f16(ar1, b1, cr);
      ci = wmma_f16(ai0, b0, ci);
      ci = wmma_f16(ai1, b1, ci);
      int col = ct * 16 + colh;
      size_t rbase = (size_t)b * 512 + col;
#pragma unroll
      for (int j = 0; j < 8; ++j) {
        int k = kt * 16 + j + 8 * hi;
        float pr = sTwR[k], pi = sTwI[k];
        float re = cr[j] * pr - ci[j] * pi;
        float im = cr[j] * pi + ci[j] * pr;
        size_t o = ((size_t)k * 56 + l) * 4096 + rbase;
        bufA_re[o] = (_Float16)re;
        bufA_im[o] = (_Float16)im;
      }
    }
  }
}

// ---------------------------------------------------------------------------
// K2: token DFT stage 2.  WG = (m, b).  Complex GEMM, scale 1/56.
// bufB[b*TOKPAD + p*56 + m][c].
// ---------------------------------------------------------------------------
__global__ void k2_stage2(const _Float16* __restrict__ bufA_re, const _Float16* __restrict__ bufA_im,
                          const _Float16* __restrict__ gWre, const _Float16* __restrict__ gWim,
                          _Float16* __restrict__ bufB_re, _Float16* __restrict__ bufB_im) {
  extern __shared__ char smem[];
  _Float16* sBre  = (_Float16*)smem;           // 64 x 512
  _Float16* sBim  = sBre + 64 * 512;
  _Float16* sWre  = sBim + 64 * 512;           // 64 x 64
  _Float16* sWim  = sWre + 64 * 64;
  _Float16* sWimN = sWim + 64 * 64;
  int tid = threadIdx.x;
  int m = blockIdx.x, b = blockIdx.y;

  for (int idx = tid; idx < 64 * 512; idx += 256) {
    int l = idx >> 9, c = idx & 511;
    _Float16 vr = (_Float16)0.0f, vi = (_Float16)0.0f;
    if (l < 56) {
      size_t o = ((size_t)m * 56 + l) * 4096 + (size_t)b * 512 + c;
      vr = bufA_re[o];
      vi = bufA_im[o];
    }
    sBre[idx] = vr;
    sBim[idx] = vi;
  }
  for (int idx = tid; idx < 64 * 64; idx += 256) {
    _Float16 wi = gWim[idx];
    sWre[idx]  = gWre[idx];
    sWim[idx]  = wi;
    sWimN[idx] = (_Float16)(-(float)wi);
  }
  __syncthreads();

  int lane = tid & 31, wave = tid >> 5;
  int colh = lane & 15, hi = lane >> 4;
  const float scale = 1.0f / 56.0f;

  for (int kt = 0; kt < 4; ++kt) {
    v16h ar0 = frag_a(sWre, 64, kt * 16, 0);
    v16h ar1 = frag_a(sWre, 64, kt * 16, 32);
    v16h ai0 = frag_a(sWim, 64, kt * 16, 0);
    v16h ai1 = frag_a(sWim, 64, kt * 16, 32);
    v16h an0 = frag_a(sWimN, 64, kt * 16, 0);
    v16h an1 = frag_a(sWimN, 64, kt * 16, 32);
    for (int i = 0; i < 4; ++i) {
      int ct = wave * 4 + i;
      v16h br0 = frag_b(sBre, 512, 0, ct * 16);
      v16h bi0 = frag_b(sBim, 512, 0, ct * 16);
      v16h br1 = frag_b(sBre, 512, 32, ct * 16);
      v16h bi1 = frag_b(sBim, 512, 32, ct * 16);
      v8f cr = zero8(), ci = zero8();
      cr = wmma_f16(ar0, br0, cr);
      cr = wmma_f16(an0, bi0, cr);
      ci = wmma_f16(ar0, bi0, ci);
      ci = wmma_f16(ai0, br0, ci);
      cr = wmma_f16(ar1, br1, cr);
      cr = wmma_f16(an1, bi1, cr);
      ci = wmma_f16(ar1, bi1, ci);
      ci = wmma_f16(ai1, br1, ci);
      int col = ct * 16 + colh;
#pragma unroll
      for (int j = 0; j < 8; ++j) {
        int p = kt * 16 + j + 8 * hi;
        size_t o = ((size_t)b * TOKPAD + (size_t)p * 56 + m) * 512 + col;
        bufB_re[o] = (_Float16)(cr[j] * scale);
        bufB_im[o] = (_Float16)(ci[j] * scale);
      }
    }
  }
}

// ---------------------------------------------------------------------------
// K3: per (token-tile, block): forward block DFT-4 prologue, complex MLP
// (relu(A*W1+b1) then A*W2+b2, softshrink).  bufC[g][blk*128+ch].
// ---------------------------------------------------------------------------
__global__ void k3_mlp(const _Float16* __restrict__ bufB_re, const _Float16* __restrict__ bufB_im,
                       const _Float16* __restrict__ gW1h, const _Float16* __restrict__ gW2h,
                       const float* __restrict__ b1, const float* __restrict__ b2,
                       _Float16* __restrict__ bufC_re, _Float16* __restrict__ bufC_im) {
  extern __shared__ char smem[];
  _Float16* sAre  = (_Float16*)smem;           // 64 x 128
  _Float16* sAim  = sAre + 64 * 128;
  _Float16* sWre  = sAim + 64 * 128;           // 128 x 128
  _Float16* sWim  = sWre + 128 * 128;
  _Float16* sWimN = sWim + 128 * 128;
  _Float16* sHre  = sWimN + 128 * 128;         // 64 x 128
  _Float16* sHim  = sHre + 64 * 128;

  int tid = threadIdx.x;
  int kb = blockIdx.y;
  size_t gbase = (size_t)blockIdx.x * 64;

  // forward block DFT: y_kb = 0.5 * sum_j (-i)^(kb*j) * x_j
  for (int idx = tid; idx < 64 * 128; idx += 256) {
    int t = idx >> 7, ch = idx & 127;
    size_t g = gbase + t;
    float yre = 0.0f, yim = 0.0f;
#pragma unroll
    for (int j = 0; j < 4; ++j) {
      size_t o = g * 512 + (size_t)j * 128 + ch;
      float xr = (float)bufB_re[o], xi = (float)bufB_im[o];
      int tt = (kb * j) & 3;
      if      (tt == 0) { yre += xr; yim += xi; }
      else if (tt == 1) { yre += xi; yim -= xr; }   // * -i
      else if (tt == 2) { yre -= xr; yim -= xi; }   // * -1
      else              { yre -= xi; yim += xr; }   // * +i
    }
    sAre[idx] = (_Float16)(0.5f * yre);
    sAim[idx] = (_Float16)(0.5f * yim);
  }
  const _Float16* w1re = gW1h + (size_t)(0 * 4 + kb) * 128 * 128;
  const _Float16* w1im = gW1h + (size_t)(1 * 4 + kb) * 128 * 128;
  for (int idx = tid; idx < 128 * 128; idx += 256) {
    _Float16 wi = w1im[idx];
    sWre[idx]  = w1re[idx];
    sWim[idx]  = wi;
    sWimN[idx] = (_Float16)(-(float)wi);
  }
  __syncthreads();

  int lane = tid & 31, wave = tid >> 5;
  int colh = lane & 15, hi = lane >> 4;

  for (int i = 0; i < 4; ++i) {
    int tp = wave * 4 + i;
    int tr = tp >> 3, ct = tp & 7;
    float brv = b1[(0 * 4 + kb) * 128 + ct * 16 + colh];
    float biv = b1[(1 * 4 + kb) * 128 + ct * 16 + colh];
    v8f cr, ci;
#pragma unroll
    for (int j = 0; j < 8; ++j) { cr[j] = brv; ci[j] = biv; }
#pragma unroll
    for (int ks = 0; ks < 4; ++ks) {
      v16h ar = frag_a(sAre, 128, tr * 16, ks * 32);
      v16h ai = frag_a(sAim, 128, tr * 16, ks * 32);
      v16h br = frag_b(sWre, 128, ks * 32, ct * 16);
      v16h bi = frag_b(sWim, 128, ks * 32, ct * 16);
      v16h bn = frag_b(sWimN, 128, ks * 32, ct * 16);
      cr = wmma_f16(ar, br, cr);
      cr = wmma_f16(ai, bn, cr);
      ci = wmma_f16(ar, bi, ci);
      ci = wmma_f16(ai, br, ci);
    }
#pragma unroll
    for (int j = 0; j < 8; ++j) {
      int row = tr * 16 + j + 8 * hi;
      int col = ct * 16 + colh;
      float r = cr[j], im = ci[j];
      sHre[row * 128 + col] = (_Float16)(r > 0.0f ? r : 0.0f);
      sHim[row * 128 + col] = (_Float16)(im > 0.0f ? im : 0.0f);
    }
  }
  __syncthreads();

  const _Float16* w2re = gW2h + (size_t)(0 * 4 + kb) * 128 * 128;
  const _Float16* w2im = gW2h + (size_t)(1 * 4 + kb) * 128 * 128;
  for (int idx = tid; idx < 128 * 128; idx += 256) {
    _Float16 wi = w2im[idx];
    sWre[idx]  = w2re[idx];
    sWim[idx]  = wi;
    sWimN[idx] = (_Float16)(-(float)wi);
  }
  __syncthreads();

  for (int i = 0; i < 4; ++i) {
    int tp = wave * 4 + i;
    int tr = tp >> 3, ct = tp & 7;
    float brv = b2[(0 * 4 + kb) * 128 + ct * 16 + colh];
    float biv = b2[(1 * 4 + kb) * 128 + ct * 16 + colh];
    v8f cr, ci;
#pragma unroll
    for (int j = 0; j < 8; ++j) { cr[j] = brv; ci[j] = biv; }
#pragma unroll
    for (int ks = 0; ks < 4; ++ks) {
      v16h ar = frag_a(sHre, 128, tr * 16, ks * 32);
      v16h ai = frag_a(sHim, 128, tr * 16, ks * 32);
      v16h br = frag_b(sWre, 128, ks * 32, ct * 16);
      v16h bi = frag_b(sWim, 128, ks * 32, ct * 16);
      v16h bn = frag_b(sWimN, 128, ks * 32, ct * 16);
      cr = wmma_f16(ar, br, cr);
      cr = wmma_f16(ai, bn, cr);
      ci = wmma_f16(ar, bi, ci);
      ci = wmma_f16(ai, br, ci);
    }
#pragma unroll
    for (int j = 0; j < 8; ++j) {
      int row = tr * 16 + j + 8 * hi;
      int col = ct * 16 + colh;
      size_t o = (gbase + row) * 512 + (size_t)kb * 128 + col;
      bufC_re[o] = (_Float16)sshrink(cr[j]);
      bufC_im[o] = (_Float16)sshrink(ci[j]);
    }
  }
}

// ---------------------------------------------------------------------------
// K4: inverse block DFT prologue + inverse token DFT stage 1 + inverse
// twiddle.  WG = (l, b).  bufD[(k*56+l)][b*512+c].
// ---------------------------------------------------------------------------
__global__ void k4_inv1(const _Float16* __restrict__ bufC_re, const _Float16* __restrict__ bufC_im,
                        const _Float16* __restrict__ gWre, const _Float16* __restrict__ gWim,
                        const float* __restrict__ TwRe, const float* __restrict__ TwIm,
                        _Float16* __restrict__ bufD_re, _Float16* __restrict__ bufD_im) {
  extern __shared__ char smem[];
  _Float16* sBre  = (_Float16*)smem;
  _Float16* sBim  = sBre + 64 * 512;
  _Float16* sWre  = sBim + 64 * 512;
  _Float16* sWim  = sWre + 64 * 64;
  _Float16* sWimN = sWim + 64 * 64;
  float*    sTwR  = (float*)(sWimN + 64 * 64);
  float*    sTwI  = sTwR + 64;
  int tid = threadIdx.x;
  int l = blockIdx.x, b = blockIdx.y;

  for (int idx = tid; idx < 64 * 512; idx += 256) {
    int m = idx >> 9, c = idx & 511;
    float yre = 0.0f, yim = 0.0f;
    if (m < 56) {
      size_t g = (size_t)b * TOKPAD + (size_t)m * 56 + l;
      int kb = c >> 7, ch = c & 127;
#pragma unroll
      for (int j = 0; j < 4; ++j) {
        size_t o = g * 512 + (size_t)j * 128 + ch;
        float xr = (float)bufC_re[o], xi = (float)bufC_im[o];
        int tt = (kb * j) & 3;
        if      (tt == 0) { yre += xr; yim += xi; }
        else if (tt == 1) { yre -= xi; yim += xr; }   // * +i
        else if (tt == 2) { yre -= xr; yim -= xi; }   // * -1
        else              { yre += xi; yim -= xr; }   // * -i
      }
      yre *= 0.5f;
      yim *= 0.5f;
    }
    sBre[idx] = (_Float16)yre;
    sBim[idx] = (_Float16)yim;
  }
  for (int idx = tid; idx < 64 * 64; idx += 256) {
    _Float16 wi = gWim[idx];
    sWre[idx]  = gWre[idx];
    sWim[idx]  = wi;
    sWimN[idx] = (_Float16)(-(float)wi);
  }
  if (tid < 64) {
    sTwR[tid] = (tid < 56) ? TwRe[tid * 56 + l] : 0.0f;
    sTwI[tid] = (tid < 56) ? TwIm[tid * 56 + l] : 0.0f;
  }
  __syncthreads();

  int lane = tid & 31, wave = tid >> 5;
  int colh = lane & 15, hi = lane >> 4;

  for (int kt = 0; kt < 4; ++kt) {
    v16h ar0 = frag_a(sWre, 64, kt * 16, 0);
    v16h ar1 = frag_a(sWre, 64, kt * 16, 32);
    v16h ai0 = frag_a(sWim, 64, kt * 16, 0);
    v16h ai1 = frag_a(sWim, 64, kt * 16, 32);
    v16h an0 = frag_a(sWimN, 64, kt * 16, 0);
    v16h an1 = frag_a(sWimN, 64, kt * 16, 32);
    for (int i = 0; i < 4; ++i) {
      int ct = wave * 4 + i;
      v16h br0 = frag_b(sBre, 512, 0, ct * 16);
      v16h bi0 = frag_b(sBim, 512, 0, ct * 16);
      v16h br1 = frag_b(sBre, 512, 32, ct * 16);
      v16h bi1 = frag_b(sBim, 512, 32, ct * 16);
      v8f cr = zero8(), ci = zero8();
      cr = wmma_f16(ar0, br0, cr);
      cr = wmma_f16(an0, bi0, cr);
      ci = wmma_f16(ar0, bi0, ci);
      ci = wmma_f16(ai0, br0, ci);
      cr = wmma_f16(ar1, br1, cr);
      cr = wmma_f16(an1, bi1, cr);
      ci = wmma_f16(ar1, bi1, ci);
      ci = wmma_f16(ai1, br1, ci);
      int col = ct * 16 + colh;
      size_t rbase = (size_t)b * 512 + col;
#pragma unroll
      for (int j = 0; j < 8; ++j) {
        int k = kt * 16 + j + 8 * hi;
        float pr = sTwR[k], pi = sTwI[k];
        float re = cr[j] * pr - ci[j] * pi;
        float im = cr[j] * pi + ci[j] * pr;
        size_t o = ((size_t)k * 56 + l) * 4096 + rbase;
        bufD_re[o] = (_Float16)re;
        bufD_im[o] = (_Float16)im;
      }
    }
  }
}

// ---------------------------------------------------------------------------
// K5: inverse token DFT stage 2, REAL part only, scale 1/56, write f32 out.
// WG = (m, b).  out[b, p*56+m, c].
// ---------------------------------------------------------------------------
__global__ void k5_inv2(const _Float16* __restrict__ bufD_re, const _Float16* __restrict__ bufD_im,
                        const _Float16* __restrict__ gWre, const _Float16* __restrict__ gWim,
                        float* __restrict__ out) {
  extern __shared__ char smem[];
  _Float16* sBre  = (_Float16*)smem;
  _Float16* sBim  = sBre + 64 * 512;
  _Float16* sWre  = sBim + 64 * 512;
  _Float16* sWimN = sWre + 64 * 64;
  int tid = threadIdx.x;
  int m = blockIdx.x, b = blockIdx.y;

  for (int idx = tid; idx < 64 * 512; idx += 256) {
    int l = idx >> 9, c = idx & 511;
    _Float16 vr = (_Float16)0.0f, vi = (_Float16)0.0f;
    if (l < 56) {
      size_t o = ((size_t)m * 56 + l) * 4096 + (size_t)b * 512 + c;
      vr = bufD_re[o];
      vi = bufD_im[o];
    }
    sBre[idx] = vr;
    sBim[idx] = vi;
  }
  for (int idx = tid; idx < 64 * 64; idx += 256) {
    sWre[idx]  = gWre[idx];
    sWimN[idx] = (_Float16)(-(float)gWim[idx]);
  }
  __syncthreads();

  int lane = tid & 31, wave = tid >> 5;
  int colh = lane & 15, hi = lane >> 4;
  const float scale = 1.0f / 56.0f;

  for (int kt = 0; kt < 4; ++kt) {
    v16h ar0 = frag_a(sWre, 64, kt * 16, 0);
    v16h ar1 = frag_a(sWre, 64, kt * 16, 32);
    v16h an0 = frag_a(sWimN, 64, kt * 16, 0);
    v16h an1 = frag_a(sWimN, 64, kt * 16, 32);
    for (int i = 0; i < 4; ++i) {
      int ct = wave * 4 + i;
      v16h br0 = frag_b(sBre, 512, 0, ct * 16);
      v16h bi0 = frag_b(sBim, 512, 0, ct * 16);
      v16h br1 = frag_b(sBre, 512, 32, ct * 16);
      v16h bi1 = frag_b(sBim, 512, 32, ct * 16);
      v8f cr = zero8();
      cr = wmma_f16(ar0, br0, cr);      // +Wre*Bre
      cr = wmma_f16(an0, bi0, cr);      // -Wim*Bim
      cr = wmma_f16(ar1, br1, cr);
      cr = wmma_f16(an1, bi1, cr);
      // p = kt*16 + j + 8*hi < 56  <=>  kt < 3 || hi == 0   (uniform per half)
      if (kt < 3 || hi == 0) {
        int col = ct * 16 + colh;
#pragma unroll
        for (int j = 0; j < 8; ++j) {
          int p = kt * 16 + j + 8 * hi;
          size_t o = ((size_t)b * 3136 + (size_t)p * 56 + m) * 512 + col;
          out[o] = cr[j] * scale;
        }
      }
    }
  }
}

// ---------------------------------------------------------------------------
extern "C" void kernel_launch(void* const* d_in, const int* in_sizes, int n_in,
                              void* d_out, int out_size, void* d_ws, size_t ws_size,
                              hipStream_t stream) {
  (void)in_sizes; (void)n_in; (void)out_size; (void)ws_size;
  const float* x  = (const float*)d_in[0];
  const float* w1 = (const float*)d_in[1];
  const float* w2 = (const float*)d_in[2];
  const float* b1 = (const float*)d_in[3];
  const float* b2 = (const float*)d_in[4];
  float* out = (float*)d_out;

  char* ws = (char*)d_ws;
  size_t off = 0;
  auto take = [&](size_t bytes) -> char* {
    char* p = ws + off;
    off = (off + bytes + 255) & ~(size_t)255;
    return p;
  };
  _Float16* Wf_re = (_Float16*)take(64 * 64 * 2);
  _Float16* Wf_im = (_Float16*)take(64 * 64 * 2);
  _Float16* Wi_re = (_Float16*)take(64 * 64 * 2);
  _Float16* Wi_im = (_Float16*)take(64 * 64 * 2);
  float* TwF_re = (float*)take(56 * 56 * 4);
  float* TwF_im = (float*)take(56 * 56 * 4);
  float* TwI_re = (float*)take(56 * 56 * 4);
  float* TwI_im = (float*)take(56 * 56 * 4);
  _Float16* gW1h = (_Float16*)take(2 * 4 * 128 * 128 * 2);
  _Float16* gW2h = (_Float16*)take(2 * 4 * 128 * 128 * 2);
  const size_t PLANE = (size_t)PROWS * 4096 * 2;  // padded f16 plane bytes
  _Float16* bufA_re = (_Float16*)take(PLANE);
  _Float16* bufA_im = (_Float16*)take(PLANE);
  _Float16* bufB_re = (_Float16*)take(PLANE);
  _Float16* bufB_im = (_Float16*)take(PLANE);
  _Float16* bufC_re = bufA_re; _Float16* bufC_im = bufA_im;   // aliases
  _Float16* bufD_re = bufB_re; _Float16* bufD_im = bufB_im;

  k0_init<<<1024, 256, 0, stream>>>(w1, w2, Wf_re, Wf_im, Wi_re, Wi_im,
                                    TwF_re, TwF_im, TwI_re, TwI_im, gW1h, gW2h);

  size_t sm1 = (size_t)(64 * 512 + 2 * 64 * 64) * 2 + 2 * 64 * 4;        // ~82 KB
  k1_stage1<<<dim3(56, 8), 256, sm1, stream>>>(x, Wf_re, Wf_im, TwF_re, TwF_im,
                                               bufA_re, bufA_im);

  size_t sm2 = (size_t)(2 * 64 * 512 + 3 * 64 * 64) * 2;                 // 152 KB
  k2_stage2<<<dim3(56, 8), 256, sm2, stream>>>(bufA_re, bufA_im, Wf_re, Wf_im,
                                               bufB_re, bufB_im);

  size_t sm3 = (size_t)(4 * 64 * 128 + 3 * 128 * 128) * 2;               // 160 KB
  k3_mlp<<<dim3(448, 4), 256, sm3, stream>>>(bufB_re, bufB_im, gW1h, gW2h,
                                             b1, b2, bufC_re, bufC_im);

  size_t sm4 = (size_t)(2 * 64 * 512 + 3 * 64 * 64) * 2 + 2 * 64 * 4;    // ~152 KB
  k4_inv1<<<dim3(56, 8), 256, sm4, stream>>>(bufC_re, bufC_im, Wi_re, Wi_im,
                                             TwI_re, TwI_im, bufD_re, bufD_im);

  size_t sm5 = (size_t)(2 * 64 * 512 + 2 * 64 * 64) * 2;                 // 144 KB
  k5_inv2<<<dim3(56, 8), 256, sm5, stream>>>(bufD_re, bufD_im, Wi_re, Wi_im, out);
}
